// EncodeProcessDecode_28114855920037
// MI455X (gfx1250) — matrix operand order repaired and checked by reading, not compiled, not run on hardware
//
#include <hip/hip_runtime.h>

// ---------------------------------------------------------------------------
// EncodeProcessDecode (MeshGraphNets-style GNN) for MI455X / gfx1250.
// All Linear layers except the decoder's 128->3 output run on v_wmma_f32_16x16x32_bf16 (bf16 in, f32 accum).
// ---------------------------------------------------------------------------

typedef __attribute__((ext_vector_type(8)))  float  v8f;
typedef __attribute__((ext_vector_type(16))) __bf16 v16bf;
typedef __attribute__((ext_vector_type(8)))  __bf16 v8bf;

#define TM 64          // rows (edges/nodes) per workgroup
#define LCH 128        // latent channels

enum { MODE_ENC = 0, MODE_EDGE = 1, MODE_NODE = 2 };

__device__ __forceinline__ __bf16 f2bf(float f) {
  unsigned u = __builtin_bit_cast(unsigned, f);
  unsigned r = (u + 0x7FFFu + ((u >> 16) & 1u)) >> 16;   // round-to-nearest-even
  return __builtin_bit_cast(__bf16, (unsigned short)r);
}

__device__ __forceinline__ float bf2f(__bf16 h) {
  unsigned u = ((unsigned)__builtin_bit_cast(unsigned short, h)) << 16;
  return __builtin_bit_cast(float, u);
}

// A fragment (16x32 bf16, ISA 7.12.2): lanes 0-15 row M=lane hold K {k0+0..7, k0+16..23},
// lanes 16-31 row M=lane-16 hold K {k0+8..15, k0+24..31}.  Two 16B LDS loads.
__device__ __forceinline__ v16bf load_a_frag(const __bf16* base, int ld, int m0,
                                             int k0, int lane) {
  const __bf16* p = base + (m0 + (lane & 15)) * ld + k0 + ((lane >> 4) << 3);
  v8bf lo = *(const v8bf*)(p);
  v8bf hi = *(const v8bf*)(p + 16);
  return __builtin_shufflevector(lo, hi, 0,1,2,3,4,5,6,7,8,9,10,11,12,13,14,15);
}

// B fragment (32x16 bf16): weights pre-transposed [Nout][Kpad]. Lanes 0-15 col=lane
// K {k0+0..7, k0+16..23}, lanes 16-31 col=lane-16 K {k0+8..15, k0+24..31}.
__device__ __forceinline__ v16bf load_b_frag(const __bf16* Wt, int ldw, int n0,
                                             int k0, int lane) {
  const __bf16* p = Wt + (n0 + (lane & 15)) * ldw + k0 + ((lane >> 4) << 3);
  v8bf lo = *(const v8bf*)(p);
  v8bf hi = *(const v8bf*)(p + 16);
  return __builtin_shufflevector(lo, hi, 0,1,2,3,4,5,6,7,8,9,10,11,12,13,14,15);
}

// One wave computes a 16x64 output strip: 4 accumulator tiles, A reused x4.
template <int K>
__device__ __forceinline__ void wave_gemm(const __bf16* A, int lda,
                                          const __bf16* Wt, int m0, int n0,
                                          int lane, v8f acc[4]) {
  for (int k0 = 0; k0 < K; k0 += 32) {
    if (k0 + 32 < K)  // stream next weight chunk (global_prefetch_b8)
      __builtin_prefetch(Wt + (n0 + (lane & 15)) * K + k0 + 32, 0, 1);
    v16bf a = load_a_frag(A, lda, m0, k0, lane);
#pragma unroll
    for (int i = 0; i < 4; ++i) {
      v16bf b = load_b_frag(Wt, K, n0 + 16 * i, k0, lane);
      acc[i] = __builtin_amdgcn_wmma_f32_16x16x32_bf16(
          false, a, false, b, (short)0, acc[i], false, false);
    }
  }
}

// ---------------------------------------------------------------------------
// Fused gather/concat -> Linear -> ReLU -> Linear -> LayerNorm -> epilogue.
//   MODE_ENC : X = pad(src[:, :Kin]) ; dst = LN(...)
//   MODE_EDGE: X = [x[snd] | x[rcv] | e] ; e += LN(...) ; atomicAdd(agg[rcv], LN)
//   MODE_NODE: X = [x | agg]            ; x += LN(...)
// ---------------------------------------------------------------------------
template <int K1, int MODE>
__global__ void __launch_bounds__(256)
mlp_block_kernel(const float* __restrict__ src, int Kin,
                 const float* __restrict__ xf, float* __restrict__ ef,
                 const int* __restrict__ senders, const int* __restrict__ receivers,
                 const float* __restrict__ aggIn, float* __restrict__ aggOut,
                 const __bf16* __restrict__ W1t, const float* __restrict__ b1,
                 const __bf16* __restrict__ W2t, const float* __restrict__ b2,
                 const float* __restrict__ lng, const float* __restrict__ lnb,
                 float* __restrict__ dst, int rows) {
  extern __shared__ char smem[];
  constexpr int XBYTES =
      (TM * K1 * 2 > TM * LCH * 4) ? TM * K1 * 2 : TM * LCH * 4;
  __bf16* Xs = (__bf16*)smem;             // [TM, K1]  (GEMM1 input, bf16)
  float*  Ys = (float*)smem;              // [TM, 128] (GEMM2 output, overlays Xs)
  __bf16* Hs = (__bf16*)(smem + XBYTES);  // [TM, 128] (hidden, bf16)

  const int row0 = blockIdx.x * TM;
  const int tid  = threadIdx.x;

  // ---- stage input tile (gather + concat + fp32->bf16) ----
  for (int idx = tid; idx < TM * K1; idx += 256) {
    const int r = idx / K1, c = idx - r * K1;
    const int gr = row0 + r;
    float v = 0.f;
    if (gr < rows) {
      if (MODE == MODE_ENC) {
        v = (c < Kin) ? src[gr * Kin + c] : 0.f;
      } else if (MODE == MODE_EDGE) {
        if (c < LCH)            v = xf[senders[gr]   * LCH + c];
        else if (c < 2 * LCH)   v = xf[receivers[gr] * LCH + (c - LCH)];
        else                    v = ef[gr * LCH + (c - 2 * LCH)];
      } else {  // MODE_NODE
        v = (c < LCH) ? xf[gr * LCH + c] : aggIn[gr * LCH + (c - LCH)];
      }
    }
    Xs[idx] = f2bf(v);
  }
  __syncthreads();

  const int lane = tid & 31;
  const int wave = tid >> 5;
  const int m0 = (wave & 3) * 16;        // row tile
  const int n0 = (wave >> 2) * 64;       // col strip
  const int rb = m0 + ((lane >> 4) << 3);  // D layout: M = v + 8*(lane>=16)
  const int cl = lane & 15;                //            N = lane & 15

  // ---- GEMM1: hidden = relu(X @ W1 + b1) -> bf16 LDS ----
  {
    v8f acc[4] = {};
    wave_gemm<K1>(Xs, K1, W1t, m0, n0, lane, acc);
#pragma unroll
    for (int i = 0; i < 4; ++i) {
      const int col = n0 + 16 * i + cl;
      const float bb = b1[col];
#pragma unroll
      for (int vv = 0; vv < 8; ++vv)
        Hs[(rb + vv) * LCH + col] = f2bf(fmaxf(acc[i][vv] + bb, 0.f));
    }
  }
  __syncthreads();  // Hs complete; Xs dead -> Ys may overlay it

  // ---- GEMM2: y = hidden @ W2 + b2 -> f32 LDS ----
  {
    v8f acc[4] = {};
    wave_gemm<LCH>(Hs, LCH, W2t, m0, n0, lane, acc);
#pragma unroll
    for (int i = 0; i < 4; ++i) {
      const int col = n0 + 16 * i + cl;
      const float bb = b2[col];
#pragma unroll
      for (int vv = 0; vv < 8; ++vv)
        Ys[(rb + vv) * LCH + col] = acc[i][vv] + bb;
    }
  }
  __syncthreads();

  // ---- LayerNorm + epilogue: 4 lanes per row, 32 cols each, shfl reduce ----
  {
    const int r = tid >> 2, part = tid & 3;
    const int gr = row0 + r;
    const float* yr = Ys + r * LCH;
    const int c0 = part * 32;

    float s = 0.f;
    for (int c = c0; c < c0 + 32; ++c) s += yr[c];
    s += __shfl_xor(s, 1); s += __shfl_xor(s, 2);
    const float mean = s * (1.f / 128.f);

    float s2 = 0.f;
    for (int c = c0; c < c0 + 32; ++c) { float d = yr[c] - mean; s2 += d * d; }
    s2 += __shfl_xor(s2, 1); s2 += __shfl_xor(s2, 2);
    const float rinv = rsqrtf(s2 * (1.f / 128.f) + 1e-5f);

    if (gr < rows) {
      const int recv = (MODE == MODE_EDGE) ? receivers[gr] : 0;
      for (int c = c0; c < c0 + 32; ++c) {
        const float u = (yr[c] - mean) * rinv * lng[c] + lnb[c];
        if (MODE == MODE_ENC) {
          dst[gr * LCH + c] = u;
        } else if (MODE == MODE_EDGE) {
          ef[gr * LCH + c] += u;                       // e = e + e_upd
          atomicAdd(&aggOut[recv * LCH + c], u);       // segment_sum(e_upd)
        } else {
          dst[gr * LCH + c] += u;                      // x = x + x_upd
        }
      }
    }
  }
}

// ---------------------------------------------------------------------------
// Decoder: Linear(128->128, WMMA) -> ReLU -> Linear(128->3, scalar) (no LN)
// ---------------------------------------------------------------------------
__global__ void __launch_bounds__(256)
decoder_kernel(const float* __restrict__ xf,
               const __bf16* __restrict__ W1t, const float* __restrict__ b1,
               const float* __restrict__ W2, const float* __restrict__ b2,
               float* __restrict__ out, int rows) {
  extern __shared__ char smem[];
  __bf16* Xs = (__bf16*)smem;                    // [TM,128]
  __bf16* Hs = (__bf16*)(smem + TM * LCH * 2);   // [TM,128]
  const int row0 = blockIdx.x * TM, tid = threadIdx.x;

  for (int idx = tid; idx < TM * LCH; idx += 256) {
    const int r = idx >> 7, c = idx & 127, gr = row0 + r;
    Xs[idx] = f2bf(gr < rows ? xf[gr * LCH + c] : 0.f);
  }
  __syncthreads();

  const int lane = tid & 31, wave = tid >> 5;
  const int m0 = (wave & 3) * 16, n0 = (wave >> 2) * 64;
  const int rb = m0 + ((lane >> 4) << 3), cl = lane & 15;

  v8f acc[4] = {};
  wave_gemm<LCH>(Xs, LCH, W1t, m0, n0, lane, acc);
#pragma unroll
  for (int i = 0; i < 4; ++i) {
    const int col = n0 + 16 * i + cl;
    const float bb = b1[col];
#pragma unroll
    for (int vv = 0; vv < 8; ++vv)
      Hs[(rb + vv) * LCH + col] = f2bf(fmaxf(acc[i][vv] + bb, 0.f));
  }
  __syncthreads();

  if (tid < TM * 3) {
    const int r = tid / 3, o = tid - r * 3, gr = row0 + r;
    if (gr < rows) {
      float s = b2[o];
      for (int k = 0; k < LCH; ++k) s += bf2f(Hs[r * LCH + k]) * W2[k * 3 + o];
      out[gr * 3 + o] = s;
    }
  }
}

// W [K][Nout] fp32  ->  Wt [Nout][Kpad] bf16 (zero-padded K), for B fragments.
__global__ void cast_wt_kernel(const float* __restrict__ W, __bf16* __restrict__ Wt,
                               int K, int Nout, int Kpad) {
  const int idx = blockIdx.x * 256 + threadIdx.x;
  if (idx >= Nout * Kpad) return;
  const int n = idx / Kpad, k = idx - n * Kpad;
  Wt[idx] = f2bf(k < K ? W[k * Nout + n] : 0.f);
}

__global__ void zero_kernel(float* __restrict__ p, int n) {
  const int i = blockIdx.x * 256 + threadIdx.x;
  if (i < n) p[i] = 0.f;
}

// ---------------------------------------------------------------------------
extern "C" void kernel_launch(void* const* d_in, const int* in_sizes, int n_in,
                              void* d_out, int out_size, void* d_ws, size_t ws_size,
                              hipStream_t stream) {
  (void)n_in; (void)out_size; (void)ws_size;

  const float* node_features = (const float*)d_in[0];
  const float* edge_features = (const float*)d_in[1];
  const int*   senders       = (const int*)d_in[2];
  const int*   receivers     = (const int*)d_in[3];
  const float* enc_node_w1 = (const float*)d_in[4];
  const float* enc_node_b1 = (const float*)d_in[5];
  const float* enc_node_w2 = (const float*)d_in[6];
  const float* enc_node_b2 = (const float*)d_in[7];
  const float* enc_node_lg = (const float*)d_in[8];
  const float* enc_node_lb = (const float*)d_in[9];
  const float* enc_edge_w1 = (const float*)d_in[10];
  const float* enc_edge_b1 = (const float*)d_in[11];
  const float* enc_edge_w2 = (const float*)d_in[12];
  const float* enc_edge_b2 = (const float*)d_in[13];
  const float* enc_edge_lg = (const float*)d_in[14];
  const float* enc_edge_lb = (const float*)d_in[15];
  const float* pe_w1 = (const float*)d_in[16];
  const float* pe_b1 = (const float*)d_in[17];
  const float* pe_w2 = (const float*)d_in[18];
  const float* pe_b2 = (const float*)d_in[19];
  const float* pe_lg = (const float*)d_in[20];
  const float* pe_lb = (const float*)d_in[21];
  const float* pn_w1 = (const float*)d_in[22];
  const float* pn_b1 = (const float*)d_in[23];
  const float* pn_w2 = (const float*)d_in[24];
  const float* pn_b2 = (const float*)d_in[25];
  const float* pn_lg = (const float*)d_in[26];
  const float* pn_lb = (const float*)d_in[27];
  const float* dec_w1 = (const float*)d_in[28];
  const float* dec_b1 = (const float*)d_in[29];
  const float* dec_w2 = (const float*)d_in[30];
  const float* dec_b2 = (const float*)d_in[31];

  const int N = in_sizes[0] / 12;
  const int E = in_sizes[1] / 7;
  const int S = 5;

  // workspace carve-up (all chunks are multiples of 256B)
  char* ws = (char*)d_ws;
  size_t off = 0;
  auto carve = [&](size_t bytes) {
    char* p = ws + off;
    off += (bytes + 255) & ~size_t(255);
    return p;
  };
  float*  xf  = (float*)carve((size_t)N * LCH * 4);
  float*  ef  = (float*)carve((size_t)E * LCH * 4);
  float*  agg = (float*)carve((size_t)N * LCH * 4);
  __bf16* enc_node_w1t = (__bf16*)carve(128 * 32 * 2);
  __bf16* enc_node_w2t = (__bf16*)carve(128 * 128 * 2);
  __bf16* enc_edge_w1t = (__bf16*)carve(128 * 32 * 2);
  __bf16* enc_edge_w2t = (__bf16*)carve(128 * 128 * 2);
  __bf16* pe_w1t = (__bf16*)carve((size_t)S * 128 * 384 * 2);
  __bf16* pe_w2t = (__bf16*)carve((size_t)S * 128 * 128 * 2);
  __bf16* pn_w1t = (__bf16*)carve((size_t)S * 128 * 256 * 2);
  __bf16* pn_w2t = (__bf16*)carve((size_t)S * 128 * 128 * 2);
  __bf16* dec_w1t = (__bf16*)carve(128 * 128 * 2);

  auto castW = [&](const float* W, __bf16* Wt, int K, int Nout, int Kpad) {
    const int n = Nout * Kpad;
    cast_wt_kernel<<<(n + 255) / 256, 256, 0, stream>>>(W, Wt, K, Nout, Kpad);
  };
  castW(enc_node_w1, enc_node_w1t, 12, 128, 32);
  castW(enc_node_w2, enc_node_w2t, 128, 128, 128);
  castW(enc_edge_w1, enc_edge_w1t, 7, 128, 32);
  castW(enc_edge_w2, enc_edge_w2t, 128, 128, 128);
  castW(dec_w1, dec_w1t, 128, 128, 128);
  for (int s = 0; s < S; ++s) {
    castW(pe_w1 + (size_t)s * 384 * 128, pe_w1t + (size_t)s * 128 * 384, 384, 128, 384);
    castW(pe_w2 + (size_t)s * 128 * 128, pe_w2t + (size_t)s * 128 * 128, 128, 128, 128);
    castW(pn_w1 + (size_t)s * 256 * 128, pn_w1t + (size_t)s * 128 * 256, 256, 128, 256);
    castW(pn_w2 + (size_t)s * 128 * 128, pn_w2t + (size_t)s * 128 * 128, 128, 128, 128);
  }

  const int gbN = (N + TM - 1) / TM;
  const int gbE = (E + TM - 1) / TM;
  constexpr size_t ldsEnc  = (size_t)TM * LCH * 4 + TM * LCH * 2;  // 48 KB
  constexpr size_t ldsEdge = (size_t)TM * 384 * 2 + TM * LCH * 2;  // 64 KB
  constexpr size_t ldsNode = (size_t)TM * LCH * 4 + TM * LCH * 2;  // 48 KB
  constexpr size_t ldsDec  = (size_t)TM * LCH * 2 * 2;             // 32 KB

  // ---- encoders ----
  mlp_block_kernel<32, MODE_ENC><<<gbN, 256, ldsEnc, stream>>>(
      node_features, 12, nullptr, nullptr, nullptr, nullptr, nullptr, nullptr,
      enc_node_w1t, enc_node_b1, enc_node_w2t, enc_node_b2,
      enc_node_lg, enc_node_lb, xf, N);
  mlp_block_kernel<32, MODE_ENC><<<gbE, 256, ldsEnc, stream>>>(
      edge_features, 7, nullptr, nullptr, nullptr, nullptr, nullptr, nullptr,
      enc_edge_w1t, enc_edge_b1, enc_edge_w2t, enc_edge_b2,
      enc_edge_lg, enc_edge_lb, ef, E);

  // ---- processor: 5 GraphNet blocks ----
  for (int s = 0; s < S; ++s) {
    zero_kernel<<<(N * LCH + 255) / 256, 256, 0, stream>>>(agg, N * LCH);
    mlp_block_kernel<384, MODE_EDGE><<<gbE, 256, ldsEdge, stream>>>(
        nullptr, 0, xf, ef, senders, receivers, nullptr, agg,
        pe_w1t + (size_t)s * 128 * 384, pe_b1 + s * 128,
        pe_w2t + (size_t)s * 128 * 128, pe_b2 + s * 128,
        pe_lg + s * 128, pe_lb + s * 128, nullptr, E);
    mlp_block_kernel<256, MODE_NODE><<<gbN, 256, ldsNode, stream>>>(
        nullptr, 0, xf, nullptr, nullptr, nullptr, agg, nullptr,
        pn_w1t + (size_t)s * 128 * 256, pn_b1 + s * 128,
        pn_w2t + (size_t)s * 128 * 128, pn_b2 + s * 128,
        pn_lg + s * 128, pn_lb + s * 128, xf, N);
  }

  // ---- decoder ----
  decoder_kernel<<<gbN, 256, ldsDec, stream>>>(xf, dec_w1t, dec_b1, dec_w2,
                                               dec_b2, (float*)d_out, N);
}
